// AnyLoc_34050500723394
// MI455X (gfx1250) — matrix operand order repaired and checked by reading, not compiled, not run on hardware
//
#include <hip/hip_runtime.h>
#include <hip/hip_bf16.h>

// Problem constants (match the reference)
#define BATCH 32
#define NPTS  1024
#define KC    64
#define DD    1536
#define EPSF  1e-12f

typedef __attribute__((ext_vector_type(16))) __bf16 v16bf;
typedef __attribute__((ext_vector_type(8)))  float  v8f;
typedef __attribute__((ext_vector_type(4)))  unsigned int u32x4;
typedef __attribute__((ext_vector_type(8)))  int i32x8;
typedef __attribute__((ext_vector_type(4)))  int i32x4;

// ---------------- workspace layout (bytes) ----------------
// [0, 196608)                 : centersN bf16  (KC*DD)
// [196608, 327680)            : labels int32   (BATCH*NPTS)
// [327680, 12910592)          : sum_desc f32   (BATCH*KC*DD)
// [12910592, 12918784)        : counts f32     (BATCH*KC)
// [12918784, 12918912)        : gsum f32       (BATCH)
#define OFF_CENTN   0
#define OFF_LABELS  196608
#define OFF_SUMD    327680
#define OFF_COUNTS  12910592
#define OFF_GSUM    12918784
#define ZERO_BYTES  (12918912 - OFF_SUMD)

// -------- block-wide sum reduction (256 threads = 8 waves, wave32) --------
__device__ __forceinline__ float block_reduce_sum(float v, float* sred) {
    #pragma unroll
    for (int off = 16; off > 0; off >>= 1) v += __shfl_xor(v, off, 32);
    const int wid = threadIdx.x >> 5;
    if ((threadIdx.x & 31) == 0) sred[wid] = v;
    __syncthreads();
    float t = 0.f;
    #pragma unroll
    for (int i = 0; i < 8; ++i) t += sred[i];
    __syncthreads();
    return t;
}

// ---------------- kernel 1: normalize centers -> bf16 ----------------
__global__ __launch_bounds__(256)
void centers_norm_kernel(const float* __restrict__ centers, __bf16* __restrict__ centersN) {
    __shared__ float sred[8];
    const int k = blockIdx.x;
    const int tid = threadIdx.x;
    const float* row = centers + (size_t)k * DD;
    float s = 0.f;
    #pragma unroll
    for (int j = 0; j < DD / 256; ++j) { float v = row[tid + 256 * j]; s += v * v; }
    const float tot = block_reduce_sum(s, sred);
    const float inv = 1.0f / fmaxf(sqrtf(tot), EPSF);
    #pragma unroll
    for (int j = 0; j < DD / 256; ++j) {
        centersN[(size_t)k * DD + tid + 256 * j] = (__bf16)(row[tid + 256 * j] * inv);
    }
}

// ---------------- kernel 2: WMMA label pass ----------------
// grid = (NPTS/128, BATCH), block = 256 (8 waves x 16 rows). Dynamic LDS: KC*DD bf16.
__global__ __launch_bounds__(256)
void labels_kernel(const float* __restrict__ descs, const __bf16* __restrict__ centersN,
                   int* __restrict__ labels) {
    extern __shared__ char smem_raw[];
    __bf16* sB = (__bf16*)smem_raw;          // [KC][DD] bf16, row = cluster

    const int b    = blockIdx.y;
    const int tid  = threadIdx.x;
    const int wave = tid >> 5;

    // ---- Stage all normalized centers into LDS (196608 B) via the Tensor Data Mover ----
#if __has_builtin(__builtin_amdgcn_tensor_load_to_lds)
    if (wave == 0) {
        // D# per CDNA5 ISA ch.8: 2D tensor, tile == whole tensor (contiguous copy).
        const unsigned lds_base = (unsigned)(uintptr_t)sB;     // low 32 bits = LDS offset
        const unsigned long long ga = (unsigned long long)(uintptr_t)centersN;
        u32x4 g0 = {};
        g0[0] = 1u;                                            // count=1 (valid user D#)
        g0[1] = lds_base;                                      // lds_addr
        g0[2] = (unsigned)(ga & 0xffffffffu);                  // global_addr[31:0]
        g0[3] = (unsigned)((ga >> 32) & 0x01ffffffu)           // global_addr[56:32]
              | (2u << 30);                                    // type=2 ("image")
        i32x8 g1 = {};
        g1[0] = (int)(1u << 16);                               // data_size=1 -> 2 bytes (bf16)
        g1[1] = (int)((unsigned)DD << 16);                     // tensor_dim0[15:0] @ bits[63:48]
        g1[2] = (int)((unsigned)KC << 16);                     // tensor_dim1[15:0] @ bits[95:80]
        g1[3] = (int)((unsigned)DD << 16);                     // tile_dim0 @ bits[127:112]
        g1[4] = KC;                                            // tile_dim1 @ bits[143:128]
        g1[5] = DD;                                            // tensor_dim0_stride @ bits[207:160]
        i32x4 g2 = {}, g3 = {};                                // unused (2D tensor)
        i32x8 g4 = {};                                         // unused trailing group (clang-23 6-arg form)
        __builtin_amdgcn_tensor_load_to_lds(g0, g1, g2, g3, g4, 0);
        __builtin_amdgcn_s_wait_tensorcnt(0);                  // s_wait_tensorcnt 0
    }
#else
    {
        const float4* src = (const float4*)centersN;
        float4* dst = (float4*)sB;
        for (int i = tid; i < (KC * DD * 2) / 16; i += 256) dst[i] = src[i];
    }
#endif
    __syncthreads();

    const int lane = tid & 31;
    const int hi   = lane >> 4;    // lane half (0/1)
    const int l16  = lane & 15;
    const int m0   = blockIdx.x * 128 + wave * 16;   // first row of this wave's tile
    const float* arow = descs + ((size_t)b * NPTS + (m0 + l16)) * DD;

    v8f acc0 = {}, acc1 = {}, acc2 = {}, acc3 = {};

    for (int k0 = 0; k0 < DD; k0 += 32) {
        __builtin_prefetch(arow + k0 + 128, 0, 3);   // global_prefetch_b8 (near-cache)

        // A fragment (16x32, contiguous K order, same as B): lane half selects K sub-block.
        // halves 0..15 <- K = k0 + hi*16 + j (contiguous)
        const int dbase = k0 + hi * 16;
        v16bf a;
        #pragma unroll
        for (int j = 0; j < 8; ++j) {
            a[j]     = (__bf16)arow[dbase + j];
            a[8 + j] = (__bf16)arow[dbase + 8 + j];
        }

        // B fragments: col = c*16 + l16, halves 0..15 <- K = k0 + hi*16 + j (contiguous)
        const __bf16* bbase = sB + k0 + hi * 16;
        v16bf b0 = *(const v16bf*)(bbase + (size_t)(0  + l16) * DD);
        v16bf b1 = *(const v16bf*)(bbase + (size_t)(16 + l16) * DD);
        v16bf b2 = *(const v16bf*)(bbase + (size_t)(32 + l16) * DD);
        v16bf b3 = *(const v16bf*)(bbase + (size_t)(48 + l16) * DD);

        acc0 = __builtin_amdgcn_wmma_f32_16x16x32_bf16(false, a, false, b0, (short)0, acc0, false, false);
        acc1 = __builtin_amdgcn_wmma_f32_16x16x32_bf16(false, a, false, b1, (short)0, acc1, false, false);
        acc2 = __builtin_amdgcn_wmma_f32_16x16x32_bf16(false, a, false, b2, (short)0, acc2, false, false);
        acc3 = __builtin_amdgcn_wmma_f32_16x16x32_bf16(false, a, false, b3, (short)0, acc3, false, false);
    }

    // C layout: lane L, VGPR r -> row M = r + 8*hi, cluster col = c*16 + l16.
    // Per-lane argmax across the 4 column tiles, then xor-shuffle reduce over the 16-lane half.
    int bi[8];
    #pragma unroll
    for (int r = 0; r < 8; ++r) {
        float v = acc0[r]; int idx = l16;
        if (acc1[r] > v) { v = acc1[r]; idx = 16 + l16; }
        if (acc2[r] > v) { v = acc2[r]; idx = 32 + l16; }
        if (acc3[r] > v) { v = acc3[r]; idx = 48 + l16; }
        #pragma unroll
        for (int off = 1; off < 16; off <<= 1) {
            float ov = __shfl_xor(v, off, 32);
            int   oi = __shfl_xor(idx, off, 32);
            if (ov > v || (ov == v && oi < idx)) { v = ov; idx = oi; }
        }
        bi[r] = idx;
    }
    if (l16 == 0) {
        #pragma unroll
        for (int r = 0; r < 8; ++r)
            labels[(size_t)b * NPTS + m0 + 8 * hi + r] = bi[r];
    }
}

// ---------------- kernel 3: normalize rows + scatter-accumulate ----------------
// grid = (NPTS, BATCH), block = 256. One block per descriptor row.
__global__ __launch_bounds__(256)
void accumulate_kernel(const float* __restrict__ descs, const int* __restrict__ labels,
                       float* __restrict__ sum_desc, float* __restrict__ counts) {
    __shared__ float sred[8];
    const int b = blockIdx.y, n = blockIdx.x, tid = threadIdx.x;
    const float* row = descs + ((size_t)b * NPTS + n) * DD;
    float v[DD / 256];
    float s = 0.f;
    #pragma unroll
    for (int j = 0; j < DD / 256; ++j) { v[j] = row[tid + 256 * j]; s += v[j] * v[j]; }
    const float tot = block_reduce_sum(s, sred);
    const float inv = 1.0f / fmaxf(sqrtf(tot), EPSF);
    const int lab = labels[(size_t)b * NPTS + n];
    float* dst = sum_desc + ((size_t)b * KC + lab) * DD;
    #pragma unroll
    for (int j = 0; j < DD / 256; ++j) atomicAdd(dst + tid + 256 * j, v[j] * inv);
    if (tid == 0) atomicAdd(counts + b * KC + lab, 1.0f);
}

// ---------------- kernel 4a: residual + intra-norm, per-batch sumsq ----------------
// grid = (KC, BATCH), block = 256.
__global__ __launch_bounds__(256)
void finalize1_kernel(const float* __restrict__ sum_desc, const float* __restrict__ counts,
                      const float* __restrict__ centers, float* __restrict__ out,
                      float* __restrict__ gsum) {
    __shared__ float sred[8];
    const int b = blockIdx.y, k = blockIdx.x, tid = threadIdx.x;
    const float cnt = counts[b * KC + k];
    const float* s   = sum_desc + ((size_t)b * KC + k) * DD;
    const float* cen = centers + (size_t)k * DD;
    float u[DD / 256];
    float ss = 0.f;
    #pragma unroll
    for (int j = 0; j < DD / 256; ++j) {
        u[j] = s[tid + 256 * j] - cnt * cen[tid + 256 * j];
        ss += u[j] * u[j];
    }
    const float tot = block_reduce_sum(ss, sred);
    const float inv = 1.0f / fmaxf(sqrtf(tot), EPSF);
    #pragma unroll
    for (int j = 0; j < DD / 256; ++j)
        out[((size_t)b * KC + k) * DD + tid + 256 * j] = u[j] * inv;
    if (tid == 0) atomicAdd(gsum + b, tot * inv * inv);
}

// ---------------- kernel 4b: final global L2 scale ----------------
__global__ __launch_bounds__(256)
void finalize2_kernel(float* __restrict__ out, const float* __restrict__ gsum) {
    const size_t idx = (size_t)blockIdx.x * 256 + threadIdx.x;
    const int b = (int)(idx / (KC * DD));
    const float g = gsum[b];
    out[idx] *= 1.0f / fmaxf(sqrtf(g), EPSF);
}

// ---------------- host launcher ----------------
extern "C" void kernel_launch(void* const* d_in, const int* in_sizes, int n_in,
                              void* d_out, int out_size, void* d_ws, size_t ws_size,
                              hipStream_t stream) {
    const float* descs   = (const float*)d_in[0];   // [B,N,D]
    const float* centers = (const float*)d_in[1];   // [K,D]
    float* out = (float*)d_out;                     // [B, K*D]

    char* ws = (char*)d_ws;
    __bf16* centersN = (__bf16*)(ws + OFF_CENTN);
    int*    labels   = (int*)(ws + OFF_LABELS);
    float*  sum_desc = (float*)(ws + OFF_SUMD);
    float*  counts   = (float*)(ws + OFF_COUNTS);
    float*  gsum     = (float*)(ws + OFF_GSUM);

    // Zero accumulation buffers (atomics below accumulate into them every call).
    (void)hipMemsetAsync(ws + OFF_SUMD, 0, ZERO_BYTES, stream);

    centers_norm_kernel<<<KC, 256, 0, stream>>>(centers, centersN);

    const size_t lds_bytes = (size_t)KC * DD * sizeof(__bf16);   // 196608 B
    labels_kernel<<<dim3(NPTS / 128, BATCH), 256, lds_bytes, stream>>>(descs, centersN, labels);

    accumulate_kernel<<<dim3(NPTS, BATCH), 256, 0, stream>>>(descs, labels, sum_desc, counts);

    finalize1_kernel<<<dim3(KC, BATCH), 256, 0, stream>>>(sum_desc, counts, centers, out, gsum);

    finalize2_kernel<<<(BATCH * KC * DD) / 256, 256, 0, stream>>>(out, gsum);
}
